// MiddleLayerEncoder_42820823941130
// MI455X (gfx1250) — matrix operand order repaired and checked by reading, not compiled, not run on hardware
//
#include <hip/hip_runtime.h>

// ---------------------------------------------------------------------------
// MI455X (gfx1250, wave32) implementation.
// Heavy GEMMs use v_wmma_f32_16x16x32_f16; segment_max via u32 atomicMax on
// float bit patterns (valid: post-ReLU values are >= 0, buffers zero-inited).
// Weights pre-packed once per launch into B-fragment lane layout so each lane
// fetches its 16 halfs with one contiguous 32B load (L2-resident, 192MB L2).
// ---------------------------------------------------------------------------

typedef __attribute__((ext_vector_type(16))) _Float16 v16h;
typedef __attribute__((ext_vector_type(8)))  _Float16 v8h;
typedef __attribute__((ext_vector_type(8)))  float    v8f;

#define NPTS 262144
#define NCLU 4096

__device__ __forceinline__ v8f wmma_f16(v16h a, v16h b, v8f c) {
  return __builtin_amdgcn_wmma_f32_16x16x32_f16(
      /*neg_a=*/false, a, /*neg_b=*/false, b,
      /*c_mod=*/(short)0, c, /*reuse_a=*/false, /*reuse_b=*/false);
}

// A fragment (16x32 f16) from an LDS row-major tile.
// lane M = lane&15; halfs 0..7 -> K = kt*32 + g + 0..7,
// halfs 8..15 -> K = kt*32 + 16 + g + 0..7, g = 8*(lane>=16).
__device__ __forceinline__ v16h load_a_frag_lds(const _Float16* tile, int stride,
                                                int kt, int lane) {
  const int row = lane & 15;
  const int g   = (lane >> 4) << 3;          // 0 or 8
  const _Float16* p = tile + row * stride + kt * 32;
  v8h lo = *(const v8h*)(p + g);             // ds_load_b128
  v8h hi = *(const v8h*)(p + 16 + g);        // ds_load_b128
  v16h a;
#pragma unroll
  for (int i = 0; i < 8; ++i) { a[i] = lo[i]; a[i + 8] = hi[i]; }
  return a;
}

// B fragment: packed contiguously per (nt, kt, lane): 16 halfs = 32 bytes.
__device__ __forceinline__ v16h load_b_frag(const _Float16* wp, int nt, int kt,
                                            int KT, int lane) {
  return *(const v16h*)(wp + ((((nt * KT) + kt) * 32 + lane) << 4));
}

// ---------------------------------------------------------------------------
__global__ void zero_kernel(unsigned* __restrict__ p, int n) {
  int i = blockIdx.x * blockDim.x + threadIdx.x;
  if (i < n) p[i] = 0u;
}

// Pack row-major W[K, ldN] (f32) into f16 B-fragment layout:
// dst[((nt*KT+kt)*32 + lane)*16 + h] = W[(kt*32 + 16*(h>=8) + 8*(lane>=16) + (h&7))*ldN + nt*16 + (lane&15)]
__global__ void pack_kernel(const float* __restrict__ W, _Float16* __restrict__ dst,
                            int K, int ldN, int KT, int total) {
  int idx = blockIdx.x * blockDim.x + threadIdx.x;
  if (idx >= total) return;
  int h    = idx & 15;
  int lane = (idx >> 4) & 31;
  int t    = idx >> 9;
  int kt   = t % KT;
  int nt   = t / KT;
  int k = kt * 32 + ((h >> 3) << 4) + ((lane >> 4) << 3) + (h & 7);
  int n = nt * 16 + (lane & 15);
  dst[idx] = (k < K) ? (_Float16)W[k * ldN + n] : (_Float16)0.f;
}

// ---------------------------------------------------------------------------
// Per-point encoder MLP 3 -> 32 -> 64 (ReLU) + per-cluster segment max.
// ~1.1 GFLOP total: plain VALU, weights staged in LDS.
__global__ __launch_bounds__(256) void encoder_kernel(
    const float* __restrict__ rel, const int* __restrict__ cluster,
    const float* __restrict__ eW1, const float* __restrict__ eb1,
    const float* __restrict__ eW2, const float* __restrict__ eb2,
    unsigned* __restrict__ neigh) {
  __shared__ float sW1[96], sb1[32], sW2[2048], sb2[64];
  const int tid = threadIdx.x;
  for (int i = tid; i < 96;   i += 256) sW1[i] = eW1[i];
  for (int i = tid; i < 32;   i += 256) sb1[i] = eb1[i];
  for (int i = tid; i < 2048; i += 256) sW2[i] = eW2[i];
  for (int i = tid; i < 64;   i += 256) sb2[i] = eb2[i];
  __syncthreads();

  const int p = blockIdx.x * 256 + tid;
  const float x0 = rel[p * 3 + 0], x1 = rel[p * 3 + 1], x2 = rel[p * 3 + 2];
  float h1[32];
#pragma unroll
  for (int j = 0; j < 32; ++j)
    h1[j] = fmaxf(sb1[j] + x0 * sW1[j] + x1 * sW1[32 + j] + x2 * sW1[64 + j], 0.f);

  const int c = cluster[p];
  unsigned* dst = neigh + (size_t)c * 64;
  for (int j = 0; j < 64; ++j) {
    float s = sb2[j];
#pragma unroll
    for (int k = 0; k < 32; ++k) s += h1[k] * sW2[k * 64 + j];
    s = fmaxf(s, 0.f);
    atomicMax(dst + j, __float_as_uint(s));   // monotone for s >= 0
  }
}

// ---------------------------------------------------------------------------
// Fused: concat[rel|feat|gather(neigh)] -> fc1(131->128, WMMA) -> ReLU
//        -> fc2(128->128, WMMA) -> ReLU -> segment-max atomics into pool.
// Block = 256 threads = 8 waves; each wave owns a 16-point row tile.
__global__ __launch_bounds__(256) void mainfc_kernel(
    const float* __restrict__ rel, const float* __restrict__ feat,
    const int* __restrict__ cluster, const float* __restrict__ neighF,
    const _Float16* __restrict__ wp1, const float* __restrict__ b1,
    const _Float16* __restrict__ wp2, const float* __restrict__ b2,
    unsigned* __restrict__ pool) {
  __shared__ _Float16 lA[8][16 * 160];   // 40 KB: per-wave 16 x 160 f16 tile
  const int tid  = threadIdx.x;
  const int w    = tid >> 5;
  const int lane = tid & 31;
  const int pBase = blockIdx.x * 128 + w * 16;
  _Float16* tile = lA[w];

  // Stage A rows: [rel(3) | feat(64) | gathered neigh_enc(64) | zero pad -> 160]
  for (int idx = lane; idx < 16 * 160; idx += 32) {
    int row = idx / 160;
    int col = idx - row * 160;
    int p = pBase + row;
    float v = 0.f;
    if (col < 3)        v = rel[p * 3 + col];
    else if (col < 67)  v = feat[p * 64 + (col - 3)];
    else if (col < 131) v = neighF[(size_t)cluster[p] * 64 + (col - 67)];
    tile[idx] = (_Float16)v;
  }
  __syncthreads();

  // Cache all fc1 A fragments (5 k-tiles of 32) before the tile is reused.
  v16h aF[5];
#pragma unroll
  for (int kt = 0; kt < 5; ++kt) aF[kt] = load_a_frag_lds(tile, 160, kt, lane);

  const int nloc = lane & 15;
  const int mhi  = (lane >> 4) << 3;     // C-layout row offset: 0 or 8

  // fc1: 8 n-tiles x 5 k-tiles of v_wmma_f32_16x16x32_f16
#pragma unroll
  for (int nt = 0; nt < 8; ++nt) {
    v8f acc = {};
#pragma unroll
    for (int kt = 0; kt < 5; ++kt)
      acc = wmma_f16(aF[kt], load_b_frag(wp1, nt, kt, 5, lane), acc);
    const int n = nt * 16 + nloc;
    const float bias = b1[n];
#pragma unroll
    for (int r = 0; r < 8; ++r) {
      float v = fmaxf(acc[r] + bias, 0.f);
      tile[(r + mhi) * 160 + n] = (_Float16)v;   // h1 back into LDS (A is dead)
    }
  }
  __syncthreads();

  // fc2 A fragments from h1 (4 k-tiles).
  v16h a2F[4];
#pragma unroll
  for (int kt = 0; kt < 4; ++kt) a2F[kt] = load_a_frag_lds(tile, 160, kt, lane);

  int cl[8];
#pragma unroll
  for (int r = 0; r < 8; ++r) cl[r] = cluster[pBase + r + mhi];

  // fc2 + ReLU + per-cluster segment max (u32 atomic max of f32 bits).
#pragma unroll
  for (int nt = 0; nt < 8; ++nt) {
    v8f acc = {};
#pragma unroll
    for (int kt = 0; kt < 4; ++kt)
      acc = wmma_f16(a2F[kt], load_b_frag(wp2, nt, kt, 4, lane), acc);
    const int n = nt * 16 + nloc;
    const float bias = b2[n];
#pragma unroll
    for (int r = 0; r < 8; ++r) {
      float v = fmaxf(acc[r] + bias, 0.f);
      atomicMax(&pool[(size_t)cl[r] * 128 + n], __float_as_uint(v));
    }
  }
}

// ---------------------------------------------------------------------------
// Global MLP over clusters: [C,128] -> 128 (ReLU) -> 256 (ReLU), all WMMA.
__global__ __launch_bounds__(256) void globalfc_kernel(
    const float* __restrict__ poolF,
    const _Float16* __restrict__ gp1, const float* __restrict__ gb1,
    const _Float16* __restrict__ gp2, const float* __restrict__ gb2,
    float* __restrict__ out) {
  __shared__ _Float16 lA[8][16 * 128];   // 32 KB
  const int tid  = threadIdx.x;
  const int w    = tid >> 5;
  const int lane = tid & 31;
  const int cBase = blockIdx.x * 128 + w * 16;
  _Float16* tile = lA[w];

  for (int idx = lane; idx < 16 * 128; idx += 32)
    tile[idx] = (_Float16)poolF[(size_t)(cBase + (idx >> 7)) * 128 + (idx & 127)];
  __syncthreads();

  v16h aF[4];
#pragma unroll
  for (int kt = 0; kt < 4; ++kt) aF[kt] = load_a_frag_lds(tile, 128, kt, lane);

  const int nloc = lane & 15;
  const int mhi  = (lane >> 4) << 3;

#pragma unroll
  for (int nt = 0; nt < 8; ++nt) {
    v8f acc = {};
#pragma unroll
    for (int kt = 0; kt < 4; ++kt)
      acc = wmma_f16(aF[kt], load_b_frag(gp1, nt, kt, 4, lane), acc);
    const int n = nt * 16 + nloc;
    const float bias = gb1[n];
#pragma unroll
    for (int r = 0; r < 8; ++r)
      tile[(r + mhi) * 128 + n] = (_Float16)fmaxf(acc[r] + bias, 0.f);
  }
  __syncthreads();

  v16h a2F[4];
#pragma unroll
  for (int kt = 0; kt < 4; ++kt) a2F[kt] = load_a_frag_lds(tile, 128, kt, lane);

  for (int nt = 0; nt < 16; ++nt) {
    v8f acc = {};
#pragma unroll
    for (int kt = 0; kt < 4; ++kt)
      acc = wmma_f16(a2F[kt], load_b_frag(gp2, nt, kt, 4, lane), acc);
    const int n = nt * 16 + nloc;
    const float bias = gb2[n];
#pragma unroll
    for (int r = 0; r < 8; ++r)
      out[(size_t)(cBase + r + mhi) * 256 + n] = fmaxf(acc[r] + bias, 0.f);
  }
}

// ---------------------------------------------------------------------------
extern "C" void kernel_launch(void* const* d_in, const int* in_sizes, int n_in,
                              void* d_out, int out_size, void* d_ws, size_t ws_size,
                              hipStream_t stream) {
  (void)in_sizes; (void)n_in; (void)out_size; (void)ws_size;
  const float* rel  = (const float*)d_in[0];
  const float* feat = (const float*)d_in[1];
  const int*   clu  = (const int*)  d_in[2];
  const float* eW1  = (const float*)d_in[3];
  const float* eb1  = (const float*)d_in[4];
  const float* eW2  = (const float*)d_in[5];
  const float* eb2  = (const float*)d_in[6];
  const float* W1   = (const float*)d_in[7];
  const float* b1   = (const float*)d_in[8];
  const float* W2   = (const float*)d_in[9];
  const float* b2   = (const float*)d_in[10];
  const float* G1   = (const float*)d_in[11];
  const float* gb1  = (const float*)d_in[12];
  const float* G2   = (const float*)d_in[13];
  const float* gb2  = (const float*)d_in[14];
  float* out = (float*)d_out;

  // Workspace layout (≈3.2 MB):
  //   neigh [C*64 u32] | pool [C*128 u32] | wp1 | wp2 | gp1 | gp2 (f16 packs)
  char* ws = (char*)d_ws;
  unsigned* neigh = (unsigned*)ws;
  unsigned* pool  = (unsigned*)(ws + (size_t)NCLU * 64 * 4);
  size_t off = (size_t)NCLU * 64 * 4 + (size_t)NCLU * 128 * 4;
  _Float16* wp1 = (_Float16*)(ws + off); off += (size_t)8  * 5 * 512 * 2;
  _Float16* wp2 = (_Float16*)(ws + off); off += (size_t)8  * 4 * 512 * 2;
  _Float16* gp1 = (_Float16*)(ws + off); off += (size_t)8  * 4 * 512 * 2;
  _Float16* gp2 = (_Float16*)(ws + off);

  // 1) zero the two segment-max accumulators (neigh + pool are contiguous)
  const int zn = NCLU * 64 + NCLU * 128;
  zero_kernel<<<(zn + 255) / 256, 256, 0, stream>>>(neigh, zn);

  // 2) pack weights into f16 B-fragment layout (tiny, once per launch)
  pack_kernel<<<(20480 + 255) / 256, 256, 0, stream>>>(W1, wp1, 131, 128, 5, 20480);
  pack_kernel<<<(16384 + 255) / 256, 256, 0, stream>>>(W2, wp2, 128, 128, 4, 16384);
  pack_kernel<<<(16384 + 255) / 256, 256, 0, stream>>>(G1, gp1, 128, 128, 4, 16384);
  pack_kernel<<<(32768 + 255) / 256, 256, 0, stream>>>(G2, gp2, 128, 256, 4, 32768);

  // 3) per-point encoder + segment max
  encoder_kernel<<<NPTS / 256, 256, 0, stream>>>(rel, clu, eW1, eb1, eW2, eb2, neigh);

  // 4) fused concat + fc1 + fc2 + segment max (the heavy WMMA kernel)
  mainfc_kernel<<<NPTS / 128, 256, 0, stream>>>(rel, feat, clu, (const float*)neigh,
                                                wp1, b1, wp2, b2, pool);

  // 5) global MLP over clusters -> output [C,256]
  globalfc_kernel<<<NCLU / 128, 256, 0, stream>>>((const float*)pool,
                                                  gp1, gb1, gp2, gb2, out);
}
